// SolutionCalculatorSurface_49435073577133
// MI455X (gfx1250) — matrix operand order repaired and checked by reading, not compiled, not run on hardware
//
#include <hip/hip_runtime.h>
#include <hip/hip_bf16.h>

// ---------------- problem constants ----------------
#define NPTS   50000
#define KNBR   6
#define PENT   7              // K+1 entries per point
#define VOUT   4
#define MROWS  (NPTS*PENT)    // 350000 = 16 * 21875
#define NTILES (MROWS/16)     // 21875 (odd)
#define NPAIRS ((NTILES+1)/2) // 10938
#define D_GEO  128
#define D_POS  32
#define D_PAR  32
#define H1     256
#define H2     128
#define D_AGG  448            // H1 + D_POS + D_GEO + D_PAR
#define K1PAD  32             // D_IN=7 padded to 32

// workspace layout (bytes)
constexpr size_t OFF_PE = 0;                                  // 32 f32
constexpr size_t OFF_W1 = 256;                                // V*256*32  f16
constexpr size_t OFF_W2 = OFF_W1 + (size_t)VOUT*256*32*2;     // V*256*256 f16
constexpr size_t OFF_A1 = OFF_W2 + (size_t)VOUT*256*256*2;    // V*256*448 f16
constexpr size_t OFF_A2 = OFF_A1 + (size_t)VOUT*256*448*2;    // V*128*256 f16
constexpr size_t OFF_P  = OFF_A2 + (size_t)VOUT*128*256*2;    // V*MROWS   f32

typedef __attribute__((ext_vector_type(16))) _Float16 v16h;
typedef __attribute__((ext_vector_type(8)))  _Float16 v8h;
typedef __attribute__((ext_vector_type(8)))  float    v8f;

// ---------------- helpers ----------------
__device__ __forceinline__ v8f wmma_f16(v16h a, v16h b, v8f c) {
  return __builtin_amdgcn_wmma_f32_16x16x32_f16(false, a, false, b, (short)0, c,
                                                false, false);
}

// A-matrix fragment (16x32 f16) from row-major LDS activations.
// lane L: row = L&15; K = 16*(L>>4) + 0..15 (contiguous)
__device__ __forceinline__ v16h load_afrag(const _Float16* act, int stride, int k0,
                                           int l16, int half) {
  const _Float16* p0 = act + l16 * stride + k0 + 16 * half;
  v8h c0 = *(const v8h*)(p0);
  v8h c1 = *(const v8h*)(p0 + 8);
  return __builtin_shufflevector(c0, c1, 0,1,2,3,4,5,6,7,8,9,10,11,12,13,14,15);
}

// B-matrix fragment (32x16 f16) from packed weights wt[col][Kpad] (col-contiguous-K).
// lane L: col = L&15; K = 16*(L>>4) + 0..15 (contiguous 32 bytes)
__device__ __forceinline__ v16h load_bfrag(const _Float16* wt, int Kpad, int ntile,
                                           int k0, int l16, int half) {
  return *(const v16h*)(wt + (size_t)(ntile * 16 + l16) * Kpad + k0 + 16 * half);
}

// D (16x16 f32): VGPR r -> row = r + 8*(L>>4), col = L&15. bias+relu -> LDS f16 row-major.
__device__ __forceinline__ void store_relu_f16(v8f c, float b, int cbase,
                                               _Float16* dst, int stride,
                                               int l16, int half) {
#pragma unroll
  for (int r = 0; r < 8; r++) {
    float x = c[r] + b;
    x = x > 0.f ? x : 0.f;
    dst[(r + 8 * half) * stride + cbase + l16] = (_Float16)x;
  }
}

// ---------------- preprocess: pack weights f32(K,N) -> f16 dst[v][n][Kpad] (zero-pad K)
__global__ void pack_w(const float* __restrict__ src, _Float16* __restrict__ dst,
                       int Kd, int Nd, int Kpad, int total) {
  int t = blockIdx.x * 256 + threadIdx.x;
  if (t >= total) return;
  int k = t % Kpad;
  int rem = t / Kpad;
  int ncol = rem % Nd;
  int v = rem / Nd;
  float x = (k < Kd) ? src[((size_t)v * Kd + k) * Nd + ncol] : 0.f;
  dst[t] = (_Float16)x;
}

// param_enc = relu((vals/refs) @ W + b), constant over N
__global__ void param_enc_k(const float* __restrict__ vals, const float* __restrict__ refs,
                            const float* __restrict__ W, const float* __restrict__ b,
                            float* __restrict__ pe) {
  int j = threadIdx.x;
  if (j < 32) {
    float s = b[j];
    for (int i = 0; i < 2; i++) s += (vals[i] / refs[i]) * W[i * 32 + j];
    pe[j] = s > 0.f ? s : 0.f;
  }
}

// ---- phase A: fused MLP chain. Block = 4 waves sharing two 16-row tiles for one v.
//      Waves split the output-column (N) range; B fragments are loaded exactly once
//      per block and each feeds two WMMAs (tile pair).
__global__ __launch_bounds__(128)
void mlp_tile_kernel(const float* __restrict__ centers, const float* __restrict__ enc_g,
                     const float* __restrict__ enc_nd, const float* __restrict__ nbrs,
                     const float* __restrict__ normals, const float* __restrict__ nnorm,
                     const float* __restrict__ areas, const float* __restrict__ nareas,
                     const float* __restrict__ pe,
                     const _Float16* __restrict__ w1t, const float* __restrict__ b1,
                     const _Float16* __restrict__ w2t, const float* __restrict__ b2,
                     const _Float16* __restrict__ a1t, const float* __restrict__ ab1,
                     const _Float16* __restrict__ a2t, const float* __restrict__ ab2,
                     const float* __restrict__ w3,  const float* __restrict__ b3,
                     float* __restrict__ partial) {
  __shared__ __align__(32) _Float16 volA[2][16 * 40];   // 16 x K1PAD, stride 40
  __shared__ __align__(32) _Float16 hbuf[2][16 * 264];  // 16 x 256, stride 264
  __shared__ __align__(32) _Float16 abuf[2][16 * 456];  // 16 x 448, stride 456
  // f32 a2 (2 x 16 x 132) is overlaid on abuf (dead by aggW2)

  const int tp   = blockIdx.x;
  const int v    = blockIdx.y;
  const int tid  = threadIdx.x;
  const int lane = tid & 31;
  const int wv   = tid >> 5;        // wave 0..3
  const int l16  = lane & 15;
  const int half = lane >> 4;
  const bool has1 = (tp * 2 + 1) < NTILES;

  // ---- build vol rows: tid 0-15 -> tile0 rows, tid 16-31 -> tile1 rows ----
  if (tid < 32) {
    const int t = half;
    const bool ok = (t == 0) || has1;
    float f[7] = {0.f, 0.f, 0.f, 0.f, 0.f, 0.f, 0.f};
    if (ok) {
      int gr = (tp * 2 + t) * 16 + l16;
      int n = gr / PENT, p = gr % PENT;
      if (p == 0) {
        f[0] = centers[n * 3 + 0]; f[1] = centers[n * 3 + 1]; f[2] = centers[n * 3 + 2];
        f[3] = normals[n * 3 + 0]; f[4] = normals[n * 3 + 1]; f[5] = normals[n * 3 + 2];
        f[6] = __logf(areas[n]) * 0.1f;
      } else {
        int base = (n * KNBR + (p - 1)) * 3;
        f[0] = nbrs[base + 0] + 1e-6f; f[1] = nbrs[base + 1] + 1e-6f; f[2] = nbrs[base + 2] + 1e-6f;
        f[3] = nnorm[base + 0] + 1e-6f; f[4] = nnorm[base + 1] + 1e-6f; f[5] = nnorm[base + 2] + 1e-6f;
        f[6] = __logf(nareas[n * KNBR + (p - 1)]) * 0.1f + 1e-6f;
      }
    }
    _Float16* row = volA[t] + l16 * 40;
#pragma unroll
    for (int k = 0; k < 7; k++) row[k] = (_Float16)f[k];
#pragma unroll
    for (int k = 7; k < 32; k++) row[k] = (_Float16)0.f;
  }
  __syncthreads();

  // ---- layer1: 7(pad32) -> 256; wave wv computes nt = wv*4 .. wv*4+3 ----
  {
    v16h a0 = load_afrag(volA[0], 40, 0, l16, half);
    v16h a1 = load_afrag(volA[1], 40, 0, l16, half);
    const _Float16* w = w1t + (size_t)v * 256 * K1PAD;
    const float* bb = b1 + v * 256;
    for (int nt = wv * 4; nt < wv * 4 + 4; nt++) {
      v16h bV = load_bfrag(w, K1PAD, nt, 0, l16, half);
      float b = bb[nt * 16 + l16];
      v8f c0 = {}; c0 = wmma_f16(a0, bV, c0);
      v8f c1 = {}; c1 = wmma_f16(a1, bV, c1);
      store_relu_f16(c0, b, nt * 16, hbuf[0], 264, l16, half);
      store_relu_f16(c1, b, nt * 16, hbuf[1], 264, l16, half);
    }
  }
  __syncthreads();

  // ---- layer2: 256 -> 256 (basis) into abuf cols 0..255; A both tiles hoisted ----
  {
    v16h A0[8], A1[8];
#pragma unroll
    for (int ks = 0; ks < 8; ks++) {
      A0[ks] = load_afrag(hbuf[0], 264, ks * 32, l16, half);
      A1[ks] = load_afrag(hbuf[1], 264, ks * 32, l16, half);
    }
    const _Float16* w = w2t + (size_t)v * 256 * 256;
    const float* bb = b2 + v * 256;
    for (int nt = wv * 4; nt < wv * 4 + 4; nt++) {
      v8f c0 = {}, c1 = {};
#pragma unroll
      for (int ks = 0; ks < 8; ks++) {
        v16h bV = load_bfrag(w, 256, nt, ks * 32, l16, half);
        c0 = wmma_f16(A0[ks], bV, c0);
        c1 = wmma_f16(A1[ks], bV, c1);
      }
      float b = bb[nt * 16 + l16];
      store_relu_f16(c0, b, nt * 16, abuf[0], 456, l16, half);
      store_relu_f16(c1, b, nt * 16, abuf[1], 456, l16, half);
    }
  }

  // ---- ctx fill: abuf cols 256..447 = enc_node(32) | enc_g(128) | param_enc(32) ----
#pragma unroll
  for (int t = 0; t < 2; t++) {
    const bool ok = (t == 0) || has1;
    int lr = tid >> 3;                     // 128 threads -> 16 rows, 8 lanes per row
    int gr = (tp * 2 + t) * 16 + lr;
    int n = gr / PENT;
    _Float16* row = abuf[t] + lr * 456 + 256;
    for (int k = (tid & 7); k < 192; k += 8) {
      float x = 0.f;
      if (ok) {
        if (k < 32)       x = enc_nd[n * 32 + k];
        else if (k < 160) x = enc_g[(size_t)n * 128 + (k - 32)];
        else              x = pe[k - 160];
      }
      row[k] = (_Float16)x;
    }
  }
  __syncthreads();

  // ---- agg layer1: 448 -> 256 into hbuf; tile0 A hoisted (112 VGPR), tile1 A streamed --
  {
    v16h A0h[14];
#pragma unroll
    for (int ks = 0; ks < 14; ks++) A0h[ks] = load_afrag(abuf[0], 456, ks * 32, l16, half);
    const _Float16* w = a1t + (size_t)v * 256 * D_AGG;
    const float* bb = ab1 + v * 256;
    for (int nt = wv * 4; nt < wv * 4 + 4; nt++) {
      v8f c0 = {}, c1 = {};
#pragma unroll
      for (int ks = 0; ks < 14; ks++) {
        v16h bV = load_bfrag(w, D_AGG, nt, ks * 32, l16, half);
        c0 = wmma_f16(A0h[ks], bV, c0);
        v16h a1f = load_afrag(abuf[1], 456, ks * 32, l16, half);
        c1 = wmma_f16(a1f, bV, c1);
      }
      float b = bb[nt * 16 + l16];
      store_relu_f16(c0, b, nt * 16, hbuf[0], 264, l16, half);
      store_relu_f16(c1, b, nt * 16, hbuf[1], 264, l16, half);
    }
  }
  __syncthreads();

  // ---- agg layer2: 256 -> 128; wave wv computes nt = wv*2 .. wv*2+1 ----
  float* a2p = (float*)&abuf[0][0];  // 2 x (16*132) f32 overlaid on abuf footprint
  {
    v16h A0[8], A1[8];
#pragma unroll
    for (int ks = 0; ks < 8; ks++) {
      A0[ks] = load_afrag(hbuf[0], 264, ks * 32, l16, half);
      A1[ks] = load_afrag(hbuf[1], 264, ks * 32, l16, half);
    }
    const _Float16* w = a2t + (size_t)v * 128 * 256;
    const float* bb = ab2 + v * 128;
    for (int nt = wv * 2; nt < wv * 2 + 2; nt++) {
      v8f c0 = {}, c1 = {};
#pragma unroll
      for (int ks = 0; ks < 8; ks++) {
        v16h bV = load_bfrag(w, 256, nt, ks * 32, l16, half);
        c0 = wmma_f16(A0[ks], bV, c0);
        c1 = wmma_f16(A1[ks], bV, c1);
      }
      float b = bb[nt * 16 + l16];
#pragma unroll
      for (int r = 0; r < 8; r++) {
        float x0 = c0[r] + b; x0 = x0 > 0.f ? x0 : 0.f;
        float x1 = c1[r] + b; x1 = x1 > 0.f ? x1 : 0.f;
        a2p[0 * 2112 + (r + 8 * half) * 132 + nt * 16 + l16] = x0;
        a2p[1 * 2112 + (r + 8 * half) * 132 + nt * 16 + l16] = x1;
      }
    }
  }
  __syncthreads();

  // ---- final 128 -> 1 dot: tid 0..31, tile = tid>>4, row = tid&15 ----
  if (tid < 32 && (half == 0 || has1)) {
    const float* w3v = w3 + v * 128;
    const float* rowp = a2p + half * 2112 + l16 * 132;
    float s = b3[v];
#pragma unroll 4
    for (int k = 0; k < 128; k++) s += rowp[k] * w3v[k];
    partial[(size_t)v * MROWS + (tp * 2 + half) * 16 + l16] = s;
  }
}

// ---------------- phase B: inverse-distance combine over the 7 entries ----------------
__global__ void combine_kernel(const float* __restrict__ centers, const float* __restrict__ normals,
                               const float* __restrict__ areas, const float* __restrict__ nbrs,
                               const float* __restrict__ nnorm, const float* __restrict__ nareas,
                               const float* __restrict__ partial, float* __restrict__ out) {
  int t = blockIdx.x * 256 + threadIdx.x;
  if (t >= VOUT * NPTS) return;
  int v = t / NPTS, n = t % NPTS;
  float c[7];
  c[0] = centers[n * 3 + 0]; c[1] = centers[n * 3 + 1]; c[2] = centers[n * 3 + 2];
  c[3] = normals[n * 3 + 0]; c[4] = normals[n * 3 + 1]; c[5] = normals[n * 3 + 2];
  c[6] = __logf(areas[n]) * 0.1f;
  float wsum = 0.f, dsum = 0.f;
  for (int p = 0; p < KNBR; p++) {
    float nb[7];
    int base = (n * KNBR + p) * 3;
    nb[0] = nbrs[base + 0] + 1e-6f; nb[1] = nbrs[base + 1] + 1e-6f; nb[2] = nbrs[base + 2] + 1e-6f;
    nb[3] = nnorm[base + 0] + 1e-6f; nb[4] = nnorm[base + 1] + 1e-6f; nb[5] = nnorm[base + 2] + 1e-6f;
    nb[6] = __logf(nareas[n * KNBR + p]) * 0.1f + 1e-6f;
    float d2 = 0.f;
#pragma unroll
    for (int k = 0; k < 7; k++) { float d = c[k] - nb[k]; d2 += d * d; }
    float inv = 1.0f / __fsqrt_rn(d2);
    wsum += partial[(size_t)v * MROWS + n * PENT + (p + 1)] * inv;
    dsum += inv;
  }
  float a0 = partial[(size_t)v * MROWS + n * PENT];
  out[(size_t)v * NPTS + n] = 0.5f * a0 + 0.5f * wsum / dsum;
}

// ---------------- launch ----------------
extern "C" void kernel_launch(void* const* d_in, const int* in_sizes, int n_in,
                              void* d_out, int out_size, void* d_ws, size_t ws_size,
                              hipStream_t stream) {
  const float* centers = (const float*)d_in[0];
  const float* enc_g   = (const float*)d_in[1];
  const float* enc_nd  = (const float*)d_in[2];
  const float* nbrs    = (const float*)d_in[3];
  const float* normals = (const float*)d_in[4];
  const float* nnorm   = (const float*)d_in[5];
  const float* areas   = (const float*)d_in[6];
  const float* nareas  = (const float*)d_in[7];
  const float* gpv     = (const float*)d_in[8];
  const float* gpr     = (const float*)d_in[9];
  const float* paramW  = (const float*)d_in[10];
  const float* paramB  = (const float*)d_in[11];
  const float* bw1     = (const float*)d_in[12];
  const float* bb1     = (const float*)d_in[13];
  const float* bw2     = (const float*)d_in[14];
  const float* bb2     = (const float*)d_in[15];
  const float* aw1     = (const float*)d_in[16];
  const float* ab1     = (const float*)d_in[17];
  const float* aw2     = (const float*)d_in[18];
  const float* ab2     = (const float*)d_in[19];
  const float* aw3     = (const float*)d_in[20];
  const float* ab3     = (const float*)d_in[21];

  char* ws = (char*)d_ws;
  float*    pe  = (float*)(ws + OFF_PE);
  _Float16* w1t = (_Float16*)(ws + OFF_W1);
  _Float16* w2t = (_Float16*)(ws + OFF_W2);
  _Float16* a1t = (_Float16*)(ws + OFF_A1);
  _Float16* a2t = (_Float16*)(ws + OFF_A2);
  float*    prt = (float*)(ws + OFF_P);

  // pack weights -> f16, transposed (col-major-K) + K zero-padding
  {
    int t1 = VOUT * 256 * K1PAD;
    int t2 = VOUT * 256 * 256;
    int t3 = VOUT * 256 * D_AGG;
    int t4 = VOUT * 128 * 256;
    pack_w<<<(t1 + 255) / 256, 256, 0, stream>>>(bw1, w1t, 7,     256, K1PAD, t1);
    pack_w<<<(t2 + 255) / 256, 256, 0, stream>>>(bw2, w2t, 256,   256, 256,   t2);
    pack_w<<<(t3 + 255) / 256, 256, 0, stream>>>(aw1, a1t, D_AGG, 256, D_AGG, t3);
    pack_w<<<(t4 + 255) / 256, 256, 0, stream>>>(aw2, a2t, 256,   128, 256,   t4);
  }
  param_enc_k<<<1, 32, 0, stream>>>(gpv, gpr, paramW, paramB, pe);

  dim3 grid(NPAIRS, VOUT);
  mlp_tile_kernel<<<grid, 128, 0, stream>>>(centers, enc_g, enc_nd, nbrs, normals, nnorm,
                                            areas, nareas, pe,
                                            w1t, bb1, w2t, bb2, a1t, ab1, a2t, ab2,
                                            aw3, ab3, prt);

  int tb = VOUT * NPTS;
  combine_kernel<<<(tb + 255) / 256, 256, 0, stream>>>(centers, normals, areas, nbrs,
                                                       nnorm, nareas, prt, (float*)d_out);
}
